// MultiHeadAttention_83013127897895
// MI455X (gfx1250) — matrix operand order repaired and checked by reading, not compiled, not run on hardware
//
#include <hip/hip_runtime.h>

typedef __attribute__((ext_vector_type(16))) _Float16 v16h;
typedef __attribute__((ext_vector_type(8)))  _Float16 v8h;
typedef __attribute__((ext_vector_type(8)))  float    v8f;
typedef __attribute__((ext_vector_type(4)))  float    v4f;

#define B_  2
#define S_  2048
#define D_  1024
#define H_  16
#define DH_ 64

static __device__ __forceinline__ v8f wmma_f16(v16h a, v16h b, v8f c) {
  // D = A(16x32 f16) * B(32x16 f16) + C(16x16 f32)
  return __builtin_amdgcn_wmma_f32_16x16x32_f16(false, a, false, b, (short)0, c,
                                                false, false);
}

// ---------------------------------------------------------------------------
// Kernel 1: transpose + convert the four 1024x1024 f32 weights to f16 (W^T),
// so GEMM B-operands become contiguous 32B loads per lane.
// ---------------------------------------------------------------------------
__global__ void wt_kernel(const float* __restrict__ Wq, const float* __restrict__ Wk,
                          const float* __restrict__ Wv, const float* __restrict__ Wo,
                          _Float16* __restrict__ wt) {
  __shared__ float t[16][17];
  const int w = blockIdx.y;
  const float* W = (w == 0) ? Wq : (w == 1) ? Wk : (w == 2) ? Wv : Wo;
  _Float16* out = wt + (size_t)w * ((size_t)D_ * D_);
  const int tx = threadIdx.x & 15, ty = threadIdx.x >> 4;
  const int i = blockIdx.x & 63, j = blockIdx.x >> 6;
  t[ty][tx] = W[(size_t)(j * 16 + ty) * D_ + i * 16 + tx];
  __syncthreads();
  // WT[n][k] = W[k][n]
  out[(size_t)(i * 16 + ty) * D_ + j * 16 + tx] = (_Float16)t[tx][ty];
}

// ---------------------------------------------------------------------------
// Kernel 2: fused Q/K/V projections. One wave = one 16x16 tile, K-loop of 32.
// blockIdx.y selects matrix (0=Q scaled by 1/sqrt(dh), 1=K, 2=V stored ^T).
// ---------------------------------------------------------------------------
__global__ void proj_kernel(const float* __restrict__ Xq, const float* __restrict__ Xk,
                            const float* __restrict__ Xv,
                            const _Float16* __restrict__ WT,  // 3 transposed mats
                            _Float16* __restrict__ Qf, _Float16* __restrict__ Kf,
                            _Float16* __restrict__ VTf) {
  const int mode = blockIdx.y;
  const float* X = (mode == 0) ? Xq : (mode == 1) ? Xk : Xv;
  const _Float16* W = WT + (size_t)mode * ((size_t)D_ * D_);
  const int wave = threadIdx.x >> 5;
  const int lane = threadIdx.x & 31;
  const int tile = blockIdx.x * 8 + wave;   // 16384 tiles = 256 row x 64 col
  const int rowT = tile >> 6;
  const int colT = tile & 63;
  const int mm = lane & 15;
  const int hi = lane >> 4;

  const float* xr = X + (size_t)(rowT * 16 + mm) * D_;
  const int ncol = colT * 16 + mm;
  const _Float16* wr = W + (size_t)ncol * D_;

  v8f c = {};
  for (int kk = 0; kk < D_; kk += 32) {
    // A k order used here (16x32, same as B): lane m=l&15; elem j -> K=16*hi+j
    v4f t0 = *(const v4f*)(xr + kk + 16 * hi);
    v4f t1 = *(const v4f*)(xr + kk + 16 * hi + 4);
    v4f t2 = *(const v4f*)(xr + kk + 16 * hi + 8);
    v4f t3 = *(const v4f*)(xr + kk + 16 * hi + 12);
    v16h a;
#pragma unroll
    for (int i = 0; i < 4; ++i) {
      a[i]      = (_Float16)t0[i];
      a[4 + i]  = (_Float16)t1[i];
      a[8 + i]  = (_Float16)t2[i];
      a[12 + i] = (_Float16)t3[i];
    }
    // B k order used here (32x16, same as A): lane n=l&15; elem j -> K=16*hi+j  (contiguous in W^T row)
    v16h b = *(const v16h*)(wr + kk + 16 * hi);
    c = wmma_f16(a, b, c);
  }

  const float scale = (mode == 0) ? 0.125f : 1.0f;  // 1/sqrt(64) folded into Q
  const int h = ncol >> 6;
  const int d = ncol & 63;
#pragma unroll
  for (int r = 0; r < 8; ++r) {
    const int g = rowT * 16 + r + 8 * hi;   // C layout: row = r + 8*hi
    const int bi = g >> 11;
    const int s = g & (S_ - 1);
    const _Float16 val = (_Float16)(c[r] * scale);
    if (mode == 0)
      Qf[((size_t)(bi * H_ + h) * S_ + s) * DH_ + d] = val;
    else if (mode == 1)
      Kf[((size_t)(bi * H_ + h) * S_ + s) * DH_ + d] = val;
    else
      VTf[((size_t)(bi * H_ + h) * DH_ + d) * S_ + s] = val;  // V transposed
  }
}

// ---------------------------------------------------------------------------
// Kernel 3: flash-attention. One wave = 16-query tile of one (b,h); online
// softmax over 64 chunks of 32 keys. Key-padding mask = -1e6 (ref MASK_VALUE).
// ---------------------------------------------------------------------------
__global__ void attn_kernel(const _Float16* __restrict__ Qf,
                            const _Float16* __restrict__ Kf,
                            const _Float16* __restrict__ VTf,
                            const int* __restrict__ vlens,
                            _Float16* __restrict__ Ctx) {
  __shared__ __align__(64) _Float16 plds[8][16 * 32];  // per-wave P tile (1KB)
  const int wave = threadIdx.x >> 5;
  const int lane = threadIdx.x & 31;
  const int bh = blockIdx.x >> 4;                       // 0..31
  const int q0 = ((blockIdx.x & 15) * 8 + wave) * 16;   // 0..2032
  const int b = bh >> 4;
  const int h = bh & 15;
  const int vl = vlens[b];
  const int mm = lane & 15;
  const int hi = lane >> 4;

  // Q tile as two A operands (dh 0..31 and 32..63)
  const _Float16* qb = Qf + (size_t)bh * S_ * DH_ + (size_t)(q0 + mm) * DH_;
  v16h aq0, aq1;
  {
    v8h x0 = *(const v8h*)(qb + 16 * hi);
    v8h x1 = *(const v8h*)(qb + 8 + 16 * hi);
    v8h x2 = *(const v8h*)(qb + 32 + 16 * hi);
    v8h x3 = *(const v8h*)(qb + 40 + 16 * hi);
#pragma unroll
    for (int i = 0; i < 8; ++i) { aq0[i]=x0[i]; aq0[8+i]=x1[i]; aq1[i]=x2[i]; aq1[8+i]=x3[i]; }
  }

  v8f o0 = {}, o1 = {}, o2 = {}, o3 = {};   // 16 x 64 output accumulator
  float mrun[8], lrun[8];
#pragma unroll
  for (int r = 0; r < 8; ++r) { mrun[r] = -3.0e38f; lrun[r] = 0.0f; }

  const _Float16* kb_base = Kf + (size_t)bh * S_ * DH_;
  const _Float16* vb_base = VTf + (size_t)bh * DH_ * S_;
  _Float16* myp = plds[wave];

  for (int ch = 0; ch < S_ / 32; ++ch) {
    const int kb = ch * 32;
    const int key0 = kb + mm;
    const int key1 = key0 + 16;

    // K^T as B operands: lane n = key, K-dim = dh (contiguous in K row)
    v16h bk00 = *(const v16h*)(kb_base + (size_t)key0 * DH_ + 16 * hi);
    v16h bk01 = *(const v16h*)(kb_base + (size_t)key0 * DH_ + 32 + 16 * hi);
    v16h bk10 = *(const v16h*)(kb_base + (size_t)key1 * DH_ + 16 * hi);
    v16h bk11 = *(const v16h*)(kb_base + (size_t)key1 * DH_ + 32 + 16 * hi);

    v8f s0 = {}, s1 = {};
    s0 = wmma_f16(aq0, bk00, s0);
    s0 = wmma_f16(aq1, bk01, s0);
    s1 = wmma_f16(aq0, bk10, s1);
    s1 = wmma_f16(aq1, bk11, s1);

    const bool ok0 = key0 < vl;
    const bool ok1 = key1 < vl;
    float p0a[8], p1a[8];
#pragma unroll
    for (int r = 0; r < 8; ++r) {
      float x0 = ok0 ? s0[r] : -1.0e6f;
      float x1 = ok1 ? s1[r] : -1.0e6f;
      float cm = fmaxf(x0, x1);                 // row reduce over 16 lanes
      cm = fmaxf(cm, __shfl_xor(cm, 1, 32));
      cm = fmaxf(cm, __shfl_xor(cm, 2, 32));
      cm = fmaxf(cm, __shfl_xor(cm, 4, 32));
      cm = fmaxf(cm, __shfl_xor(cm, 8, 32));
      const float nm = fmaxf(mrun[r], cm);
      const float sc = __expf(mrun[r] - nm);
      const float p0 = __expf(x0 - nm);
      const float p1 = __expf(x1 - nm);
      float rs = p0 + p1;
      rs += __shfl_xor(rs, 1, 32);
      rs += __shfl_xor(rs, 2, 32);
      rs += __shfl_xor(rs, 4, 32);
      rs += __shfl_xor(rs, 8, 32);
      lrun[r] = lrun[r] * sc + rs;
      mrun[r] = nm;
      o0[r] *= sc; o1[r] *= sc; o2[r] *= sc; o3[r] *= sc;
      p0a[r] = p0; p1a[r] = p1;
    }

    // Re-layout P (C-layout) -> A-layout via per-wave LDS tile.
    __syncthreads();
#pragma unroll
    for (int r = 0; r < 8; ++r) {
      const int row = r + 8 * hi;
      myp[row * 32 + mm]      = (_Float16)p0a[r];
      myp[row * 32 + 16 + mm] = (_Float16)p1a[r];
    }
    __syncthreads();

    v16h pa;
    {
      v8h lo  = *(const v8h*)(myp + mm * 32 + 16 * hi);       // keys 16*hi..+8
      v8h hi8 = *(const v8h*)(myp + mm * 32 + 8 + 16 * hi);   // keys 16*hi+8..+8
#pragma unroll
      for (int i = 0; i < 8; ++i) { pa[i] = lo[i]; pa[8 + i] = hi8[i]; }
    }

    // V^T as B operands: lane n = dh col, K-dim = key (contiguous in V^T row)
    v16h bv0 = *(const v16h*)(vb_base + (size_t)(0 * 16 + mm) * S_ + kb + 16 * hi);
    v16h bv1 = *(const v16h*)(vb_base + (size_t)(1 * 16 + mm) * S_ + kb + 16 * hi);
    v16h bv2 = *(const v16h*)(vb_base + (size_t)(2 * 16 + mm) * S_ + kb + 16 * hi);
    v16h bv3 = *(const v16h*)(vb_base + (size_t)(3 * 16 + mm) * S_ + kb + 16 * hi);
    o0 = wmma_f16(pa, bv0, o0);
    o1 = wmma_f16(pa, bv1, o1);
    o2 = wmma_f16(pa, bv2, o2);
    o3 = wmma_f16(pa, bv3, o3);
  }

#pragma unroll
  for (int r = 0; r < 8; ++r) {
    const float inv = 1.0f / lrun[r];
    const size_t rowoff = ((size_t)b * S_ + q0 + r + 8 * hi) * D_ + h * DH_;
    Ctx[rowoff + 0 * 16 + mm] = (_Float16)(o0[r] * inv);
    Ctx[rowoff + 1 * 16 + mm] = (_Float16)(o1[r] * inv);
    Ctx[rowoff + 2 * 16 + mm] = (_Float16)(o2[r] * inv);
    Ctx[rowoff + 3 * 16 + mm] = (_Float16)(o3[r] * inv);
  }
}

// ---------------------------------------------------------------------------
// Kernel 4: output projection  out(f32) = Ctx(f16) @ Wo  (via Wo^T f16)
// ---------------------------------------------------------------------------
__global__ void oproj_kernel(const _Float16* __restrict__ Ctx,
                             const _Float16* __restrict__ WoT,
                             float* __restrict__ Out) {
  const int wave = threadIdx.x >> 5;
  const int lane = threadIdx.x & 31;
  const int tile = blockIdx.x * 8 + wave;
  const int rowT = tile >> 6;
  const int colT = tile & 63;
  const int mm = lane & 15;
  const int hi = lane >> 4;

  const _Float16* ar = Ctx + (size_t)(rowT * 16 + mm) * D_;
  const _Float16* wr = WoT + (size_t)(colT * 16 + mm) * D_;

  v8f c = {};
  for (int kk = 0; kk < D_; kk += 32) {
    v8h a0 = *(const v8h*)(ar + kk + 16 * hi);
    v8h a1 = *(const v8h*)(ar + kk + 8 + 16 * hi);
    v16h a;
#pragma unroll
    for (int i = 0; i < 8; ++i) { a[i] = a0[i]; a[8 + i] = a1[i]; }
    v16h b = *(const v16h*)(wr + kk + 16 * hi);
    c = wmma_f16(a, b, c);
  }
  const int col = colT * 16 + mm;
#pragma unroll
  for (int r = 0; r < 8; ++r) {
    const int g = rowT * 16 + r + 8 * hi;
    Out[(size_t)g * D_ + col] = c[r];
  }
}

// ---------------------------------------------------------------------------
extern "C" void kernel_launch(void* const* d_in, const int* in_sizes, int n_in,
                              void* d_out, int out_size, void* d_ws, size_t ws_size,
                              hipStream_t stream) {
  const float* q  = (const float*)d_in[0];
  const float* k  = (const float*)d_in[1];
  const float* v  = (const float*)d_in[2];
  const float* Wq = (const float*)d_in[3];
  const float* Wk = (const float*)d_in[4];
  const float* Wv = (const float*)d_in[5];
  const float* Wo = (const float*)d_in[6];
  const int*   vl = (const int*)d_in[7];
  float* out = (float*)d_out;

  // Workspace layout (f16 elements). Total = 20 Mi halves = 40 MB.
  _Float16* ws = (_Float16*)d_ws;
  const size_t M1 = (size_t)1 << 20;
  _Float16* Qf  = ws;             // [B][H][S][dh]   4 Mi halves
  _Float16* Kf  = ws + 4 * M1;    // [B][H][S][dh]
  _Float16* VTf = ws + 8 * M1;    // [B][H][dh][S]
  _Float16* Ctx = ws + 12 * M1;   // [B*S][D]
  _Float16* WT  = ws + 16 * M1;   // WqT,WkT,WvT (1 Mi each)
  _Float16* WoT = ws + 19 * M1;   // WoT

  wt_kernel  <<<dim3(64 * 64, 4), 256, 0, stream>>>(Wq, Wk, Wv, Wo, WT);
  proj_kernel<<<dim3(2048, 3),    256, 0, stream>>>(q, k, v, WT, Qf, Kf, VTf);
  attn_kernel<<<dim3(512),        256, 0, stream>>>(Qf, Kf, VTf, vl, Ctx);
  oproj_kernel<<<dim3(2048),      256, 0, stream>>>(Ctx, WoT, out);
}
